// Subband_Quantizer_61967788147241
// MI455X (gfx1250) — compile-verified, matrix operand order checked by reading, not run on hardware
//
#include <hip/hip_runtime.h>
#include <hip/hip_bf16.h>
#include <stdint.h>

// Problem constants (match reference)
#define G_  8
#define L_  2
#define D_  128
#define K_  1024
#define CD_ 8
#define B_  8
#define T_  4096

#define TOK_PER_BLK 256
#define KTILES (K_ / 16)          // 64

// d_out flat layout (return-order concat, all f32)
#define ZQ_OFF     0
#define CODES_OFF  33554432ULL    // B*G*D*T
#define LAT_OFF    34078720ULL    // + B*G*L*T
#define COMMIT_OFF 38273024ULL    // + B*G*L*CD*T
#define CBLOSS_OFF 38273025ULL
#define LOSS_NORM  (1.0f / 2097152.0f)   // 1/(B*G*CD*T), per-level mean

typedef _Float16 v16h __attribute__((ext_vector_type(16)));
typedef float    v8f  __attribute__((ext_vector_type(8)));
typedef unsigned int v8u __attribute__((ext_vector_type(8)));

__device__ __forceinline__ unsigned pack2h(float a, float b) {
    _Float16 ha = (_Float16)a, hb = (_Float16)b;
    unsigned short ua = __builtin_bit_cast(unsigned short, ha);
    unsigned short ub = __builtin_bit_cast(unsigned short, hb);
    return (unsigned)ua | ((unsigned)ub << 16);
}

__device__ __forceinline__ v8u vmax8(v8u a, v8u b) {
#if __has_builtin(__builtin_elementwise_max)
    return __builtin_elementwise_max(a, b);
#else
    v8u r;
    #pragma unroll
    for (int i = 0; i < 8; ++i) r[i] = a[i] > b[i] ? a[i] : b[i];
    return r;
#endif
}

// ---------------------------------------------------------------------------
// Pass 1: normalize codebook rows, pack as f16 pairs in WMMA B-matrix per-lane
// format (uint4 per code = K(c0,c1)..K(c6,c7)); also zero the loss scalars.
// ---------------------------------------------------------------------------
__global__ void __launch_bounds__(256)
cb_prep_kernel(const float* __restrict__ cb, uint4* __restrict__ cbp,
               float* __restrict__ losses) {
    int e = blockIdx.x * blockDim.x + threadIdx.x;   // (g*L + l)*K + k
    if (e == 0) { losses[0] = 0.0f; losses[1] = 0.0f; }
    if (e >= G_ * L_ * K_) return;
    const float* src = cb + (size_t)e * CD_;
    float v[CD_]; float ss = 0.0f;
    #pragma unroll
    for (int c = 0; c < CD_; ++c) { v[c] = src[c]; ss += v[c] * v[c]; }
    float inv = 1.0f / fmaxf(sqrtf(ss), 1e-12f);
    uint4 p;
    p.x = pack2h(v[0]*inv, v[1]*inv);
    p.y = pack2h(v[2]*inv, v[3]*inv);
    p.z = pack2h(v[4]*inv, v[5]*inv);
    p.w = pack2h(v[6]*inv, v[7]*inv);
    cbp[e] = p;
}

// ---------------------------------------------------------------------------
// Pass 2: fused 2-level residual VQ. One block = 256 tokens of one (b,g).
// Residual lives in VGPRs for both levels; code search via
// v_wmma_f32_16x16x32_f16 (16 tokens x 16 codes per issue, CD=8 in K=0..7,
// C operand = +2.0 bias so sims become positive -> unsigned sortable keys).
// ---------------------------------------------------------------------------
__global__ void __launch_bounds__(256)
rvq_kernel(const float* __restrict__ z,
           const float* __restrict__ in_w,  const float* __restrict__ in_b,
           const float* __restrict__ codebook,
           const float* __restrict__ out_w, const float* __restrict__ out_b,
           const uint4* __restrict__ cbp,
           float* __restrict__ out) {
    const int tid  = threadIdx.x;
    const int lane = tid & 31;
    const int wave = tid >> 5;
    const int sub  = lane & 15;
    const bool lo  = lane < 16;
    const int wbase = wave * 32;

    const int tTiles = T_ / TOK_PER_BLK;              // 16
    const int bid = blockIdx.x;
    const int tt = bid % tTiles;
    const int g  = (bid / tTiles) % G_;
    const int b  = bid / (tTiles * G_);
    const int t  = tt * TOK_PER_BLK + tid;

    __shared__ float s_w [CD_ * D_];          // in_w, transposed to [d][c]
    __shared__ float s_ow[D_ * CD_];          // out_w, native [d][c]
    __shared__ float s_b [CD_];
    __shared__ float s_ob[D_];
    __shared__ uint4 s_cb[K_];                // packed f16 codebook (16 KB)
    __shared__ uint4 s_enc[TOK_PER_BLK];      // packed f16 enc_n per token
    __shared__ unsigned s_rk[TOK_PER_BLK * 17]; // argmax key candidates
    __shared__ float s_loss[TOK_PER_BLK];

    // residual column for this token, kept in VGPRs
    float r[D_];
    const size_t zbase = ((size_t)(b * G_ + g) * D_) * T_ + t;
    #pragma unroll
    for (int d = 0; d < D_; ++d) r[d] = z[zbase + (size_t)d * T_];

    float lossAcc = 0.0f;

    #pragma unroll 1
    for (int l = 0; l < L_; ++l) {
        __syncthreads();  // protect LDS reuse across levels
        {   // stage weights/biases + packed codebook for this (g,l)
            const float* wsrc  = in_w  + (size_t)(g * L_ + l) * CD_ * D_;
            const float* owsrc = out_w + (size_t)(g * L_ + l) * D_ * CD_;
            for (int i = tid; i < CD_ * D_; i += TOK_PER_BLK) {
                int c = i >> 7, d = i & (D_ - 1);
                s_w[d * CD_ + c] = wsrc[i];       // transpose -> [d][c]
            }
            for (int i = tid; i < D_ * CD_; i += TOK_PER_BLK) s_ow[i] = owsrc[i];
            const uint4* cbsrc = cbp + (size_t)(g * L_ + l) * K_;
            for (int i = tid; i < K_; i += TOK_PER_BLK) s_cb[i] = cbsrc[i];
            if (tid < CD_) s_b[tid]  = in_b [(g * L_ + l) * CD_ + tid];
            if (tid < D_)  s_ob[tid] = out_b[(g * L_ + l) * D_  + tid];
        }
        __syncthreads();

        // z_e = in_w @ r + in_b  (8x128 matvec, f32)
        float ze[CD_];
        #pragma unroll
        for (int c = 0; c < CD_; ++c) ze[c] = s_b[c];
        const float4* w4 = (const float4*)s_w;
        #pragma unroll
        for (int d = 0; d < D_; ++d) {
            float rv = r[d];
            float4 wa = w4[d * 2], wb = w4[d * 2 + 1];
            ze[0] = fmaf(wa.x, rv, ze[0]); ze[1] = fmaf(wa.y, rv, ze[1]);
            ze[2] = fmaf(wa.z, rv, ze[2]); ze[3] = fmaf(wa.w, rv, ze[3]);
            ze[4] = fmaf(wb.x, rv, ze[4]); ze[5] = fmaf(wb.y, rv, ze[5]);
            ze[6] = fmaf(wb.z, rv, ze[6]); ze[7] = fmaf(wb.w, rv, ze[7]);
        }

        // latents output (pre-normalization, as in reference)
        {
            const size_t latBase = ((size_t)(b * (G_ * L_) + g * L_ + l) * CD_) * T_ + t;
            #pragma unroll
            for (int c = 0; c < CD_; ++c)
                out[LAT_OFF + latBase + (size_t)c * T_] = ze[c];
        }

        // normalize + pack to f16 for the WMMA search
        float ss = 0.0f;
        #pragma unroll
        for (int c = 0; c < CD_; ++c) ss += ze[c] * ze[c];
        float inv = 1.0f / fmaxf(sqrtf(ss), 1e-12f);
        uint4 ep;
        ep.x = pack2h(ze[0]*inv, ze[1]*inv);
        ep.y = pack2h(ze[2]*inv, ze[3]*inv);
        ep.z = pack2h(ze[4]*inv, ze[5]*inv);
        ep.w = pack2h(ze[6]*inv, ze[7]*inv);
        s_enc[tid] = ep;
        __syncthreads();

        // Build A matrices (16-bit A 16x32 layout): lanes 0-15 carry rows
        // M=0..15 with K=0..7 in v0..v3; K=8..31 slots zero in ALL lanes.
        // Because A's K>=8 region is zero, B's K>=8 region (lanes 16-31 and
        // v4..v7) may hold arbitrary finite data -> no zero-selects on B.
        uint4 e0 = s_enc[wbase + sub];
        uint4 e1 = s_enc[wbase + 16 + sub];
        v8u a0u = { lo ? e0.x : 0u, lo ? e0.y : 0u, lo ? e0.z : 0u, lo ? e0.w : 0u,
                    0u, 0u, 0u, 0u };
        v8u a1u = { lo ? e1.x : 0u, lo ? e1.y : 0u, lo ? e1.z : 0u, lo ? e1.w : 0u,
                    0u, 0u, 0u, 0u };
        v16h A0 = __builtin_bit_cast(v16h, a0u);
        v16h A1 = __builtin_bit_cast(v16h, a1u);

        // Running argmax over 64 code tiles via sortable unsigned keys:
        // sim+2 > 0 (C operand carries the bias), key = high22(bits) | (1023-code).
        // Vector (v8u) form throughout; 2-way unroll lets max(max(b,k),k')
        // collapse into v_max3_u32.
        const v8f cBias = { 2.0f, 2.0f, 2.0f, 2.0f, 2.0f, 2.0f, 2.0f, 2.0f };
        v8u best0 = { 0u, 0u, 0u, 0u, 0u, 0u, 0u, 0u };
        v8u best1 = { 0u, 0u, 0u, 0u, 0u, 0u, 0u, 0u };

        #pragma unroll 2
        for (int kt = 0; kt < KTILES; ++kt) {
            uint4 bq = s_cb[kt * 16 + sub];
            v8u bu = { bq.x, bq.y, bq.z, bq.w, 0u, 0u, 0u, 0u };
            v16h Bm = __builtin_bit_cast(v16h, bu);
            v8f d0 = __builtin_amdgcn_wmma_f32_16x16x32_f16(
                false, A0, false, Bm, (short)0, cBias, false, false);
            v8f d1 = __builtin_amdgcn_wmma_f32_16x16x32_f16(
                false, A1, false, Bm, (short)0, cBias, false, false);
            unsigned low = 1023u - (unsigned)(kt * 16 + sub);  // ties -> lowest code
            v8u k0 = (__builtin_bit_cast(v8u, d0) & 0xFFFFFC00u) | low;
            v8u k1 = (__builtin_bit_cast(v8u, d1) & 0xFFFFFC00u) | low;
            best0 = vmax8(best0, k0);
            best1 = vmax8(best1, k1);
        }

        // Cross-lane reduce: each lane's 8 D-rows are tokens
        // M = (lane<16 ? 0 : 8) + i of each half-tile; column = sub.
        {
            int mof = lo ? 0 : 8;
            #pragma unroll
            for (int i = 0; i < 8; ++i) {
                int t0 = wbase + mof + i;
                int t1 = wbase + 16 + mof + i;
                s_rk[t0 * 17 + sub] = best0[i];
                s_rk[t1 * 17 + sub] = best1[i];
            }
        }
        __syncthreads();
        unsigned bestKey = 0u;
        #pragma unroll
        for (int c = 0; c < 16; ++c) {
            unsigned k = s_rk[tid * 17 + c];
            bestKey = k > bestKey ? k : bestKey;
        }
        int myIdx = 1023 - (int)(bestKey & 1023u);

        // codes output (as f32)
        out[CODES_OFF + ((size_t)(b * (G_ * L_) + g * L_ + l)) * T_ + t] = (float)myIdx;

        // gather raw codebook entry; accumulate losses
        const float* ce = codebook + ((size_t)((g * L_ + l) * K_ + myIdx)) * CD_;
        float4 q0 = ((const float4*)ce)[0];
        float4 q1 = ((const float4*)ce)[1];
        float zq_[CD_] = { q0.x, q0.y, q0.z, q0.w, q1.x, q1.y, q1.z, q1.w };
        #pragma unroll
        for (int c = 0; c < CD_; ++c) {
            float dlt = ze[c] - zq_[c];
            lossAcc += dlt * dlt;   // commit == cb_loss numerically
        }

        // residual -= out_w @ z_q + out_b   (128x8 matvec, f32)
        const float4* ow4 = (const float4*)s_ow;
        #pragma unroll
        for (int d = 0; d < D_; ++d) {
            float4 wa = ow4[d * 2], wb = ow4[d * 2 + 1];
            float acc = s_ob[d];
            acc = fmaf(wa.x, zq_[0], acc); acc = fmaf(wa.y, zq_[1], acc);
            acc = fmaf(wa.z, zq_[2], acc); acc = fmaf(wa.w, zq_[3], acc);
            acc = fmaf(wb.x, zq_[4], acc); acc = fmaf(wb.y, zq_[5], acc);
            acc = fmaf(wb.z, zq_[6], acc); acc = fmaf(wb.w, zq_[7], acc);
            r[d] -= acc;
        }
    }

    // zq = z - residual_final (z reload hits the 192MB L2)
    #pragma unroll
    for (int d = 0; d < D_; ++d) {
        size_t o = zbase + (size_t)d * T_;
        out[ZQ_OFF + o] = z[o] - r[d];
    }

    // block-reduce losses, atomically accumulate scaled means
    s_loss[tid] = lossAcc;
    __syncthreads();
    for (int s = TOK_PER_BLK / 2; s > 0; s >>= 1) {
        if (tid < s) s_loss[tid] += s_loss[tid + s];
        __syncthreads();
    }
    if (tid == 0) {
        float v = s_loss[0] * LOSS_NORM;
        atomicAdd(out + COMMIT_OFF, v);
        atomicAdd(out + CBLOSS_OFF, v);
    }
}

extern "C" void kernel_launch(void* const* d_in, const int* in_sizes, int n_in,
                              void* d_out, int out_size, void* d_ws, size_t ws_size,
                              hipStream_t stream) {
    (void)in_sizes; (void)n_in; (void)out_size; (void)ws_size;
    const float* z        = (const float*)d_in[0];
    const float* in_w     = (const float*)d_in[1];
    const float* in_b     = (const float*)d_in[2];
    const float* codebook = (const float*)d_in[3];
    const float* out_w    = (const float*)d_in[4];
    const float* out_b    = (const float*)d_in[5];
    float* out = (float*)d_out;
    uint4* cbp = (uint4*)d_ws;   // G*L*K uint4 = 256 KB

    cb_prep_kernel<<<(G_ * L_ * K_ + 255) / 256, 256, 0, stream>>>(
        codebook, cbp, out + COMMIT_OFF);

    const int nBlocks = B_ * G_ * (T_ / TOK_PER_BLK);   // 1024
    rvq_kernel<<<nBlocks, TOK_PER_BLK, 0, stream>>>(
        z, in_w, in_b, codebook, out_w, out_b, cbp, out);
}
